// InvariantValuePolicy_23459111370871
// MI455X (gfx1250) — matrix operand order repaired and checked by reading, not compiled, not run on hardware
//
#include <hip/hip_runtime.h>
#include <hip/hip_bf16.h>

// ---------------------------------------------------------------------------
// InvariantValuePolicy on MI455X (gfx1250): WMMA f16->f32 MLP fwd+bwd + BoxCDQP
// ---------------------------------------------------------------------------

constexpr int STATE    = 32;
constexpr int HIDDEN   = 256;
constexpr int NOUT     = 32;
constexpr int ACT      = 8;
constexpr int QP_ITERS = 30;

constexpr int MT = 16;   // batch rows per block
constexpr int AS = 272;  // LDS row stride in halves (16B-aligned, padded)

typedef _Float16 half8_t  __attribute__((ext_vector_type(8)));
typedef _Float16 half16_t __attribute__((ext_vector_type(16)));
typedef float    f32x8    __attribute__((ext_vector_type(8)));

// Packed weight-fragment offsets in d_ws (units: halves).
// Each 32x16 B tile = 32 lanes * 16 halves = 512 halves.
constexpr int SZ_TILE   = 512;
constexpr int OFF_W0    = 0;                          // KT=1,  NT=16
constexpr int SZ_W0     = 1 * 16 * SZ_TILE;
constexpr int OFF_W1    = OFF_W0 + SZ_W0;             // KT=8,  NT=16
constexpr int SZ_W1     = 8 * 16 * SZ_TILE;
constexpr int OFF_W2    = OFF_W1 + SZ_W1;             // KT=8,  NT=16
constexpr int SZ_W2     = 8 * 16 * SZ_TILE;
constexpr int OFF_WOUT  = OFF_W2 + SZ_W2;             // KT=8,  NT=2
constexpr int SZ_WOUT   = 8 * 2 * SZ_TILE;
constexpr int OFF_WOUTT = OFF_WOUT + SZ_WOUT;         // KT=1,  NT=16
constexpr int SZ_WOUTT  = 1 * 16 * SZ_TILE;
constexpr int OFF_W2T   = OFF_WOUTT + SZ_WOUTT;       // KT=8,  NT=16
constexpr int SZ_W2T    = 8 * 16 * SZ_TILE;
constexpr int OFF_W1T   = OFF_W2T + SZ_W2T;           // KT=8,  NT=16
constexpr int SZ_W1T    = 8 * 16 * SZ_TILE;
constexpr int OFF_W0T   = OFF_W1T + SZ_W1T;           // KT=8,  NT=2
constexpr int SZ_W0T    = 8 * 2 * SZ_TILE;
constexpr int OFF_G     = OFF_W0T + SZ_W0T;           // KT=1,  NT=1 (N padded 8->16)

constexpr float LOG2E = 1.4426950408889634f;
constexpr float LN2   = 0.6931471805599453f;

// ---------------------------------------------------------------------------
// Prep: pack a row-major f32 weight (or its transpose) into per-lane WMMA
// B-fragment layout: element j of lane l in tile (kt,nt) holds
//   K = kt*32 + (j&7) + (j>=8 ? 16 : 0) + (l<16 ? 0 : 8),  N = nt*16 + (l&15)
// ---------------------------------------------------------------------------
__global__ void pack_weights_kernel(const float* __restrict__ src,
                                    _Float16* __restrict__ dst,
                                    int KT, int NT, int srcStride,
                                    int transposed, int Nsrc)
{
    int id = blockIdx.x * blockDim.x + threadIdx.x;
    int total = KT * NT * SZ_TILE;
    if (id >= total) return;
    int t    = id >> 9;        // tile index
    int w    = id & 511;
    int lane = w >> 4;
    int j    = w & 15;
    int kt   = t / NT;
    int nt   = t - kt * NT;
    int k    = kt * 32 + (j & 7) + ((j >= 8) ? 16 : 0) + ((lane < 16) ? 0 : 8);
    int n    = nt * 16 + (lane & 15);
    float v = 0.0f;
    if (n < Nsrc)
        v = transposed ? src[n * srcStride + k] : src[k * srcStride + n];
    dst[id] = (_Float16)v;
}

// ---------------------------------------------------------------------------
// One 16x16 output tile: C += A(16 x 32*KT, from LDS) * Bpacked, C init = binit
// ---------------------------------------------------------------------------
__device__ __forceinline__ f32x8 tile_mma(const _Float16* __restrict__ Ap,
                                          const _Float16* __restrict__ Bp,
                                          int KT, int NT, int nt, int lane,
                                          float binit)
{
    f32x8 acc;
#pragma unroll
    for (int r = 0; r < 8; ++r) acc[r] = binit;

    const int row  = lane & 15;
    const int off0 = (lane < 16) ? 0 : 8;

    for (int kt = 0; kt < KT; ++kt) {
        // A fragment: two 16B LDS loads per lane (documented 16-bit A layout)
        const _Float16* ab = Ap + row * AS + kt * 32 + off0;
        half8_t lo = *(const half8_t*)ab;
        half8_t hi = *(const half8_t*)(ab + 16);
        half16_t a = __builtin_shufflevector(lo, hi,
            0, 1, 2, 3, 4, 5, 6, 7, 8, 9, 10, 11, 12, 13, 14, 15);

        // B fragment: pre-packed, contiguous 32B per lane
        const _Float16* bb = Bp + (size_t)(kt * NT + nt) * SZ_TILE + lane * 16;
        half16_t b = *(const half16_t*)bb;

        if (kt + 1 < KT)  // lowers to global_prefetch_b8
            __builtin_prefetch((const void*)(Bp + (size_t)((kt + 1) * NT + nt) * SZ_TILE + lane * 16), 0, 1);

        acc = __builtin_amdgcn_wmma_f32_16x16x32_f16(
            /*neg_a=*/false, a, /*neg_b=*/false, b,
            /*c_mod=*/(short)0, acc, /*reuse_a=*/false, /*reuse_b=*/false);
    }
    return acc;
}

// Branch-free stable softplus + sigmoid sharing one exp:
//   e = exp(-|z|)            (v_exp_f32)
//   softplus = max(z,0) + log(1+e)   (v_log_f32; arg in (1,2], no fixup)
//   sigmoid  = z>=0 ? 1/(1+e) : 1 - 1/(1+e)   (v_rcp_f32 + cndmask)
__device__ __forceinline__ void store_softplus(f32x8 acc,
                                               _Float16* __restrict__ hArr,
                                               _Float16* __restrict__ sArr,
                                               int nt, int lane)
{
    const int col = nt * 16 + (lane & 15);
    const int rb  = (lane >> 4) * 8;
#pragma unroll
    for (int r = 0; r < 8; ++r) {
        float z  = acc[r];
        float e  = __builtin_amdgcn_exp2f(-__builtin_fabsf(z) * LOG2E);
        float p  = 1.0f + e;
        float t  = __builtin_amdgcn_rcpf(p);
        float h  = fmaxf(z, 0.0f) + __builtin_amdgcn_logf(p) * LN2;
        float s  = (z >= 0.0f) ? t : (1.0f - t);
        hArr[(rb + r) * AS + col] = (_Float16)h;
        sArr[(rb + r) * AS + col] = (_Float16)s;
    }
}

__device__ __forceinline__ void store_grad(f32x8 acc,
                                           const _Float16* __restrict__ sArr,
                                           _Float16* __restrict__ gArr,
                                           int nt, int lane)
{
    const int col = nt * 16 + (lane & 15);
    const int rb  = (lane >> 4) * 8;
#pragma unroll
    for (int r = 0; r < 8; ++r) {
        float g = acc[r] * (float)sArr[(rb + r) * AS + col];
        gArr[(rb + r) * AS + col] = (_Float16)g;
    }
}

// ---------------------------------------------------------------------------
// Main fused kernel: 1 block = 16 batch rows, 8 waves, full fwd+bwd+QP
// ---------------------------------------------------------------------------
__global__ __launch_bounds__(256)
void ivp_kernel(const float* __restrict__ x,
                const float* __restrict__ b0, const float* __restrict__ b1,
                const float* __restrict__ b2, const float* __restrict__ bout,
                const float* __restrict__ R,
                const float* __restrict__ lower, const float* __restrict__ upper,
                const _Float16* __restrict__ ws,
                float* __restrict__ out)
{
    __shared__ alignas(16) _Float16 actA[MT * AS];
    __shared__ alignas(16) _Float16 actB[MT * AS];
    __shared__ alignas(16) _Float16 s0[MT * AS];
    __shared__ alignas(16) _Float16 s1[MT * AS];
    __shared__ alignas(16) _Float16 s2[MT * AS];
    __shared__ alignas(16) float    cArr[16 * 16];

    const int tid  = threadIdx.x;
    const int lane = tid & 31;
    const int wave = tid >> 5;
    const int m0   = blockIdx.x * MT;

    // Load x tile (16x32 f32) -> actA as f16
    for (int i = tid; i < MT * STATE; i += blockDim.x) {
        int r = i >> 5, c = i & 31;
        actA[r * AS + c] = (_Float16)x[(m0 + r) * STATE + c];
    }
    __syncthreads();

    // ---- Forward: L0  z0 = x@W0+b0 ; h0->actB, sig(z0)->s0
    for (int nt = wave; nt < 16; nt += 8) {
        f32x8 acc = tile_mma(actA, ws + OFF_W0, 1, 16, nt, lane,
                             b0[nt * 16 + (lane & 15)]);
        store_softplus(acc, actB, s0, nt, lane);
    }
    __syncthreads();

    // ---- L1: h1->actA, sig(z1)->s1
    for (int nt = wave; nt < 16; nt += 8) {
        f32x8 acc = tile_mma(actB, ws + OFF_W1, 8, 16, nt, lane,
                             b1[nt * 16 + (lane & 15)]);
        store_softplus(acc, actA, s1, nt, lane);
    }
    __syncthreads();

    // ---- L2: h2->actB, sig(z2)->s2
    for (int nt = wave; nt < 16; nt += 8) {
        f32x8 acc = tile_mma(actA, ws + OFF_W2, 8, 16, nt, lane,
                             b2[nt * 16 + (lane & 15)]);
        store_softplus(acc, actB, s2, nt, lane);
    }
    __syncthreads();

    // ---- L3: z3 = h2@Wout+bout (16x32); g3 = relu'(z3) -> actA cols 0..31
    if (wave < 2) {
        f32x8 acc = tile_mma(actB, ws + OFF_WOUT, 8, 2, wave, lane,
                             bout[wave * 16 + (lane & 15)]);
        const int col = wave * 16 + (lane & 15);
        const int rb  = (lane >> 4) * 8;
#pragma unroll
        for (int r = 0; r < 8; ++r)
            actA[(rb + r) * AS + col] = (acc[r] > 0.0f) ? (_Float16)1.0f
                                                        : (_Float16)0.0f;
    }
    __syncthreads();

    // ---- Backward: g2 = (g3 @ Wout^T) * sig(z2) -> actB
    for (int nt = wave; nt < 16; nt += 8) {
        f32x8 acc = tile_mma(actA, ws + OFF_WOUTT, 1, 16, nt, lane, 0.0f);
        store_grad(acc, s2, actB, nt, lane);
    }
    __syncthreads();

    // ---- g1 = (g2 @ W2^T) * sig(z1) -> actA
    for (int nt = wave; nt < 16; nt += 8) {
        f32x8 acc = tile_mma(actB, ws + OFF_W2T, 8, 16, nt, lane, 0.0f);
        store_grad(acc, s1, actA, nt, lane);
    }
    __syncthreads();

    // ---- g0 = (g1 @ W1^T) * sig(z0) -> actB
    for (int nt = wave; nt < 16; nt += 8) {
        f32x8 acc = tile_mma(actA, ws + OFF_W1T, 8, 16, nt, lane, 0.0f);
        store_grad(acc, s0, actB, nt, lane);
    }
    __syncthreads();

    // ---- dvdx = g0 @ W0^T (16x32) -> actA cols 0..31 (f16)
    if (wave < 2) {
        f32x8 acc = tile_mma(actB, ws + OFF_W0T, 8, 2, wave, lane, 0.0f);
        const int col = wave * 16 + (lane & 15);
        const int rb  = (lane >> 4) * 8;
#pragma unroll
        for (int r = 0; r < 8; ++r)
            actA[(rb + r) * AS + col] = (_Float16)acc[r];
    }
    __syncthreads();

    // ---- c = dvdx @ G (G packed, N padded to 16) -> cArr f32
    if (wave == 0) {
        f32x8 acc = tile_mma(actA, ws + OFF_G, 1, 1, 0, lane, 0.0f);
        const int col = lane & 15;
        const int rb  = (lane >> 4) * 8;
#pragma unroll
        for (int r = 0; r < 8; ++r)
            cArr[(rb + r) * 16 + col] = acc[r];
    }
    __syncthreads();

    // ---- BoxCDQP: one sample per lane (16 lanes of wave 0)
    if (tid < MT) {
        float Q[ACT][ACT], invd[ACT], ci[ACT], u[ACT], lo[ACT], hi[ACT];
#pragma unroll
        for (int i = 0; i < ACT; ++i) {
#pragma unroll
            for (int j = 0; j < ACT; ++j)
                Q[i][j] = 2.0f * R[i * ACT + j];
            invd[i] = __builtin_amdgcn_rcpf(Q[i][i]);
            ci[i]   = cArr[tid * 16 + i];
            u[i]    = 0.0f;
            lo[i]   = lower[i];
            hi[i]   = upper[i];
        }
        for (int it = 0; it < QP_ITERS; ++it) {
#pragma unroll
            for (int i = 0; i < ACT; ++i) {
                float g = ci[i];
#pragma unroll
                for (int j = 0; j < ACT; ++j)
                    g += Q[i][j] * u[j];
                float v = u[i] - g * invd[i];
                u[i] = fminf(fmaxf(v, lo[i]), hi[i]);
            }
        }
#pragma unroll
        for (int i = 0; i < ACT; ++i)
            out[(size_t)(m0 + tid) * ACT + i] = u[i];
    }
}

// ---------------------------------------------------------------------------
extern "C" void kernel_launch(void* const* d_in, const int* in_sizes, int n_in,
                              void* d_out, int out_size, void* d_ws, size_t ws_size,
                              hipStream_t stream)
{
    const float* x     = (const float*)d_in[0];
    // d_in[1] = t (unused)
    const float* W0    = (const float*)d_in[2];
    const float* b0    = (const float*)d_in[3];
    const float* W1    = (const float*)d_in[4];
    const float* b1    = (const float*)d_in[5];
    const float* W2    = (const float*)d_in[6];
    const float* b2    = (const float*)d_in[7];
    const float* Wout  = (const float*)d_in[8];
    const float* bout  = (const float*)d_in[9];
    const float* G     = (const float*)d_in[10];
    const float* R     = (const float*)d_in[11];
    const float* lower = (const float*)d_in[12];
    const float* upper = (const float*)d_in[13];
    _Float16*    ws    = (_Float16*)d_ws;
    float*       out   = (float*)d_out;

    const int Btot = in_sizes[0] / STATE;

    auto pack = [&](int off, const float* src, int KT, int NT,
                    int stride, int transposed, int Nsrc) {
        int total = KT * NT * SZ_TILE;
        pack_weights_kernel<<<(total + 255) / 256, 256, 0, stream>>>(
            src, ws + off, KT, NT, stride, transposed, Nsrc);
    };

    pack(OFF_W0,    W0,   1, 16, HIDDEN, 0, HIDDEN);  // x@W0
    pack(OFF_W1,    W1,   8, 16, HIDDEN, 0, HIDDEN);  // h0@W1
    pack(OFF_W2,    W2,   8, 16, HIDDEN, 0, HIDDEN);  // h1@W2
    pack(OFF_WOUT,  Wout, 8,  2, NOUT,   0, NOUT);    // h2@Wout
    pack(OFF_WOUTT, Wout, 1, 16, NOUT,   1, HIDDEN);  // g3@Wout^T
    pack(OFF_W2T,   W2,   8, 16, HIDDEN, 1, HIDDEN);  // g2@W2^T
    pack(OFF_W1T,   W1,   8, 16, HIDDEN, 1, HIDDEN);  // g1@W1^T
    pack(OFF_W0T,   W0,   8,  2, HIDDEN, 1, STATE);   // g0@W0^T
    pack(OFF_G,     G,    1,  1, ACT,    0, ACT);     // dvdx@G (N padded)

    ivp_kernel<<<Btot / MT, 256, 0, stream>>>(
        x, b0, b1, b2, bout, R, lower, upper, ws, out);
}
